// MetaPathEncoderInstance_88441966559844
// MI455X (gfx1250) — compile-verified, matrix operand order checked by reading, not run on hardware
//
#include <hip/hip_runtime.h>

#define NN 2048
#define DD 256
#define KK 20
#define HH 4

typedef __attribute__((ext_vector_type(16))) __bf16 v16bf;
typedef __attribute__((ext_vector_type(8)))  float  v8f;

union V16 { v16bf v; uint4 q[2]; };

__device__ __forceinline__ unsigned short f32_to_bf16(float f) {
    union { float f; unsigned u; } v; v.f = f;
    unsigned r = v.u + 0x7FFFu + ((v.u >> 16) & 1u);
    return (unsigned short)(r >> 16);
}

__device__ __forceinline__ float bf16_to_f32(unsigned short u) {
    union { unsigned u; float f; } v; v.u = ((unsigned)u) << 16;
    return v.f;
}

// ---------------- wa[h][d] = sum_e W[h][d][e] * a[h][e] ----------------
__global__ void wa_kernel(const float* __restrict__ W, const float* __restrict__ a,
                          float* __restrict__ wa) {
    int i = blockIdx.x * blockDim.x + threadIdx.x;
    if (i >= HH * DD) return;
    int h = i / DD, d = i % DD;
    const float* wrow = W + (h * DD + d) * DD;
    const float* arow = a + h * DD;
    float s = 0.f;
    for (int e = 0; e < DD; e++) s += wrow[e] * arow[e];
    wa[i] = s;
}

// ------------- Wt[m][e][d] = bf16(W[m][d][e])  (transpose+convert) -------------
__global__ void wt_kernel(const float* __restrict__ W, unsigned short* __restrict__ Wt,
                          int numMat) {
    int i = blockIdx.x * blockDim.x + threadIdx.x;
    if (i >= numMat * DD * DD) return;
    int m = i / (DD * DD), r = i % (DD * DD);
    int d = r / DD, e = r % DD;
    Wt[m * DD * DD + e * DD + d] = f32_to_bf16(W[m * DD * DD + d * DD + e]);
}

// ---------------- s_hi[h][n] = X_t[n] . wa[h] ----------------
__global__ void shi_kernel(const float* __restrict__ Xt, const float* __restrict__ wa,
                           float* __restrict__ shi) {
    __shared__ float sred[256];
    int n = blockIdx.x, t = threadIdx.x;
    float x = Xt[n * DD + t];
    for (int h = 0; h < HH; h++) {
        sred[t] = x * wa[h * DD + t];
        __syncthreads();
        for (int s = 128; s > 0; s >>= 1) {
            if (t < s) sred[t] += sred[t + s];
            __syncthreads();
        }
        if (t == 0) shi[h * NN + n] = sred[0];
        __syncthreads();
    }
}

// -------- One wave per (n,k): gather+max-pool inst, store bf16 inst to ws,
//          e[h][n][k] = leakyrelu(inst . wa[h] + shi[h,n]) --------
__global__ void score_store_kernel(const float* __restrict__ t0, const float* __restrict__ t1,
                                   const float* __restrict__ t2, const float* __restrict__ t3,
                                   const float* __restrict__ t4,
                                   const int* __restrict__ idx, int P,
                                   const float* __restrict__ wa, const float* __restrict__ shi,
                                   float* __restrict__ e, unsigned short* __restrict__ instb) {
    int gw = (blockIdx.x * blockDim.x + threadIdx.x) >> 5;
    int lane = threadIdx.x & 31;
    if (gw >= NN * KK) return;
    int n = gw / KK, k = gw % KK;
    const int* ip = idx + (n * KK + k) * P;
    const float* tabs[5] = { t0, t1, t2, t3, t4 };
    float inst[8];
    {
        const float* r0 = t0 + (size_t)ip[0] * DD + lane * 8;
        float4 v0 = *(const float4*)(r0);
        float4 v1 = *(const float4*)(r0 + 4);
        inst[0] = v0.x; inst[1] = v0.y; inst[2] = v0.z; inst[3] = v0.w;
        inst[4] = v1.x; inst[5] = v1.y; inst[6] = v1.z; inst[7] = v1.w;
        for (int p = 1; p < P; p++) {
            const float* rp = tabs[p] + (size_t)ip[p] * DD + lane * 8;
            float4 w0 = *(const float4*)(rp);
            float4 w1 = *(const float4*)(rp + 4);
            inst[0] = fmaxf(inst[0], w0.x); inst[1] = fmaxf(inst[1], w0.y);
            inst[2] = fmaxf(inst[2], w0.z); inst[3] = fmaxf(inst[3], w0.w);
            inst[4] = fmaxf(inst[4], w1.x); inst[5] = fmaxf(inst[5], w1.y);
            inst[6] = fmaxf(inst[6], w1.z); inst[7] = fmaxf(inst[7], w1.w);
        }
    }
    // coalesced bf16 store of the pooled instance vector
    union { uint4 q; unsigned short u[8]; } pk;
    #pragma unroll
    for (int j = 0; j < 8; j++) pk.u[j] = f32_to_bf16(inst[j]);
    *(uint4*)(void*)(instb + (size_t)(n * KK + k) * DD + lane * 8) = pk.q;
    // attention logits
    #pragma unroll
    for (int h = 0; h < HH; h++) {
        const float* wah = wa + h * DD + lane * 8;
        float s = 0.f;
        #pragma unroll
        for (int j = 0; j < 8; j++) s += inst[j] * wah[j];
        for (int off = 16; off > 0; off >>= 1) s += __shfl_xor(s, off, 32);
        if (lane == 0) {
            float x = s + shi[h * NN + n];
            e[h * NN * KK + n * KK + k] = (x > 0.f) ? x : 0.2f * x;
        }
    }
}

// ---------------- softmax over K, in place; one thread per (h,n) ----------------
__global__ void softmax_kernel(float* __restrict__ e) {
    int i = blockIdx.x * blockDim.x + threadIdx.x;
    if (i >= HH * NN) return;
    float* row = e + (size_t)i * KK;
    float m = row[0];
    for (int k = 1; k < KK; k++) m = fmaxf(m, row[k]);
    float s = 0.f;
    for (int k = 0; k < KK; k++) { float ex = __expf(row[k] - m); row[k] = ex; s += ex; }
    float inv = 1.f / s;
    for (int k = 0; k < KK; k++) row[k] *= inv;
}

// -------- ctx[h][n][d] = sum_k alpha[h,n,k]*inst[n,k,d] (reads bf16 inst); block per n --------
__global__ void ctx_kernel(const unsigned short* __restrict__ instb,
                           const float* __restrict__ alpha,
                           unsigned short* __restrict__ ctxb) {
    __shared__ float sal[HH * KK];
    int n = blockIdx.x, t = threadIdx.x;
    if (t < HH * KK) {
        int h = t / KK, k = t % KK;
        sal[t] = alpha[(size_t)h * NN * KK + n * KK + k];
    }
    __syncthreads();
    float acc[HH] = { 0.f, 0.f, 0.f, 0.f };
    const unsigned short* ib = instb + (size_t)n * KK * DD + t;
    for (int k = 0; k < KK; k++) {
        float v = bf16_to_f32(ib[k * DD]);
        #pragma unroll
        for (int h = 0; h < HH; h++) acc[h] += sal[h * KK + k] * v;
    }
    #pragma unroll
    for (int h = 0; h < HH; h++)
        ctxb[(size_t)h * NN * DD + n * DD + t] = f32_to_bf16(acc[h]);
}

// -------- WMMA GEMM: out[N,D] = epilogue( scale * sum_m A_m[N,D] @ B_m[D,D] ) --------
// A bf16 row-major; Bt bf16 with Bt[m][e][d] = B[m][d][e] (cols contiguous).
// Optionally also emits a bf16 copy of the output (outB != nullptr).
__global__ void wmma_gemm(const unsigned short* __restrict__ A,
                          const unsigned short* __restrict__ Bt,
                          const float* __restrict__ bias,
                          float* __restrict__ out,
                          unsigned short* __restrict__ outB,
                          int numMat, float scale, int doTanh) {
    int wid  = (blockIdx.x * blockDim.x + threadIdx.x) >> 5;
    int lane = threadIdx.x & 31;
    const int tilesN = DD / 16;                 // 16
    const int totalTiles = (NN / 16) * tilesN;  // 2048
    if (wid >= totalTiles) return;
    int tm = wid / tilesN, tn = wid % tilesN;
    int rowBase = tm * 16, colBase = tn * 16;
    int r = lane & 15, hsel = lane >> 4;
    v8f c = { 0.f, 0.f, 0.f, 0.f, 0.f, 0.f, 0.f, 0.f };
    for (int m = 0; m < numMat; m++) {
        const unsigned short* Arow = A + (size_t)m * NN * DD + (size_t)(rowBase + r) * DD;
        const unsigned short* Bcol = Bt + (size_t)m * DD * DD + (size_t)(colBase + r) * DD;
        #pragma unroll
        for (int k0 = 0; k0 < DD; k0 += 32) {
            V16 a, b;
            a.q[0] = *(const uint4*)(const void*)(Arow + k0 + hsel * 8);
            a.q[1] = *(const uint4*)(const void*)(Arow + k0 + 16 + hsel * 8);
            b.q[0] = *(const uint4*)(const void*)(Bcol + k0 + hsel * 8);
            b.q[1] = *(const uint4*)(const void*)(Bcol + k0 + 16 + hsel * 8);
            c = __builtin_amdgcn_wmma_f32_16x16x32_bf16(
                    false, a.v, false, b.v, (short)0, c, false, false);
        }
    }
    int col = colBase + r;
    float bi = bias ? bias[col] : 0.f;
    #pragma unroll
    for (int v = 0; v < 8; v++) {
        int row = rowBase + v + hsel * 8;
        float val = c[v] * scale + bi;
        if (doTanh) val = tanhf(val);
        out[(size_t)row * DD + col] = val;
        if (outB) outB[(size_t)row * DD + col] = f32_to_bf16(val);
    }
}

// ---------------- sem[e] = mean_n tmp[n][e]; block per e ----------------
__global__ void mean_kernel(const float* __restrict__ tmp, float* __restrict__ sem) {
    __shared__ float sr[128];
    int e = blockIdx.x, t = threadIdx.x;
    float s = 0.f;
    for (int n = t; n < NN; n += 128) s += tmp[(size_t)n * DD + e];
    sr[t] = s;
    __syncthreads();
    for (int k = 64; k > 0; k >>= 1) {
        if (t < k) sr[t] += sr[t + k];
        __syncthreads();
    }
    if (t == 0) sem[e] = sr[0] * (1.f / NN);
}

// ------- beta = softmax_p( sum_d tanh(sem[p][d]) * qm[d] ); single block of 256 -------
__global__ void beta_kernel(const float* __restrict__ sem, const float* __restrict__ qm,
                            int Pn, float* __restrict__ betaWs, float* __restrict__ betaOut) {
    __shared__ float sr[256];
    __shared__ float sc[3];
    int t = threadIdx.x;
    for (int p = 0; p < Pn; p++) {
        sr[t] = tanhf(sem[p * DD + t]) * qm[t];
        __syncthreads();
        for (int k = 128; k > 0; k >>= 1) {
            if (t < k) sr[t] += sr[t + k];
            __syncthreads();
        }
        if (t == 0) sc[p] = sr[0];
        __syncthreads();
    }
    if (t == 0) {
        float m = sc[0];
        for (int p = 1; p < Pn; p++) m = fmaxf(m, sc[p]);
        float s = 0.f, ex[3];
        for (int p = 0; p < Pn; p++) { ex[p] = __expf(sc[p] - m); s += ex[p]; }
        for (int p = 0; p < Pn; p++) {
            float b = ex[p] / s;
            betaWs[p] = b;
            betaOut[p] = b;
        }
    }
}

// ---------------- out = sum_p beta[p] * H_p ----------------
__global__ void combine_kernel(const float* __restrict__ H0, const float* __restrict__ H1,
                               const float* __restrict__ H2, int Pn,
                               const float* __restrict__ beta, float* __restrict__ out) {
    int i = blockIdx.x * blockDim.x + threadIdx.x;
    if (i >= NN * DD) return;
    float v = beta[0] * H0[i];
    if (Pn > 1) v += beta[1] * H1[i];
    if (Pn > 2) v += beta[2] * H2[i];
    out[i] = v;
}

extern "C" void kernel_launch(void* const* d_in, const int* in_sizes, int n_in,
                              void* d_out, int out_size, void* d_ws, size_t ws_size,
                              hipStream_t stream) {
    const float* X_drug = (const float*)d_in[0];
    const float* X_dis  = (const float*)d_in[1];
    const float* X_gene = (const float*)d_in[2];
    const int* idxs[6] = { (const int*)d_in[3], (const int*)d_in[4], (const int*)d_in[5],
                           (const int*)d_in[6], (const int*)d_in[7], (const int*)d_in[8] };
    const float* Wmp[6]; const float* amp[6];
    for (int m = 0; m < 6; m++) { Wmp[m] = (const float*)d_in[9 + 2 * m]; amp[m] = (const float*)d_in[10 + 2 * m]; }

    // workspace carve-up (all 256B aligned)
    char* w = (char*)d_ws;
    auto carve = [&](size_t bytes) { char* p = w; w += (bytes + 255) & ~(size_t)255; return p; };
    float*          wa    = (float*)carve(HH * DD * 4);
    float*          shi   = (float*)carve((size_t)HH * NN * 4);
    float*          attn  = (float*)carve((size_t)HH * NN * KK * 4);
    unsigned short* Wt    = (unsigned short*)carve((size_t)HH * DD * DD * 2);
    unsigned short* instb = (unsigned short*)carve((size_t)NN * KK * DD * 2);
    unsigned short* ctxb  = (unsigned short*)carve((size_t)HH * NN * DD * 2);
    float*          Hmp   = (float*)carve((size_t)6 * NN * DD * 4);
    unsigned short* Hmpb  = (unsigned short*)carve((size_t)6 * NN * DD * 2);
    unsigned short* Wmt   = (unsigned short*)carve((size_t)DD * DD * 2);
    float*          tmp   = (float*)carve((size_t)NN * DD * 4);
    float*          sem   = (float*)carve(3 * DD * 4);
    float*          betaW = (float*)carve(256);

    struct MPDesc { int P; const float* t[5]; const float* Xt; };
    MPDesc mps[6] = {
        { 3, { X_drug, X_dis, X_drug, X_drug, X_drug }, X_drug },  // rdr
        { 5, { X_drug, X_dis, X_gene, X_dis, X_drug }, X_drug },   // rdgdr
        { 3, { X_dis, X_gene, X_dis, X_dis, X_dis },   X_dis  },   // dgd
        { 3, { X_dis, X_drug, X_dis, X_dis, X_dis },   X_dis  },   // drd
        { 4, { X_dis, X_drug, X_drug, X_dis, X_dis },  X_dis  },   // drrd
        { 5, { X_gene, X_dis, X_drug, X_dis, X_gene }, X_gene },   // gdrdg
    };

    const int ND = NN * DD;

    for (int m = 0; m < 6; m++) {
        const MPDesc& mp = mps[m];
        wa_kernel<<<(HH * DD + 255) / 256, 256, 0, stream>>>(Wmp[m], amp[m], wa);
        wt_kernel<<<(HH * DD * DD + 255) / 256, 256, 0, stream>>>(Wmp[m], Wt, HH);
        shi_kernel<<<NN, 256, 0, stream>>>(mp.Xt, wa, shi);
        score_store_kernel<<<(NN * KK * 32) / 256, 256, 0, stream>>>(
            mp.t[0], mp.t[1], mp.t[2], mp.t[3], mp.t[4], idxs[m], mp.P, wa, shi, attn, instb);
        softmax_kernel<<<(HH * NN + 255) / 256, 256, 0, stream>>>(attn);
        ctx_kernel<<<NN, 256, 0, stream>>>(instb, attn, ctxb);
        wmma_gemm<<<512, 128, 0, stream>>>(ctxb, Wt, nullptr,
                                           Hmp + (size_t)m * ND, Hmpb + (size_t)m * ND,
                                           HH, 1.0f / HH, 0);
    }

    struct Grp { int ps[3]; int np; int wmIdx; size_t outOff; size_t betaOff; };
    Grp grps[3] = {
        { { 0, 1, 0 }, 2, 21, 0,             (size_t)3 * ND + 0 },  // drug
        { { 2, 3, 4 }, 3, 24, (size_t)ND,    (size_t)3 * ND + 2 },  // dis
        { { 5, 5, 5 }, 1, 27, (size_t)2 * ND,(size_t)3 * ND + 5 },  // gene
    };
    float* outF = (float*)d_out;

    for (int g = 0; g < 3; g++) {
        const Grp& G = grps[g];
        const float* Wm = (const float*)d_in[G.wmIdx];
        const float* bm = (const float*)d_in[G.wmIdx + 1];
        const float* qm = (const float*)d_in[G.wmIdx + 2];
        wt_kernel<<<(DD * DD + 255) / 256, 256, 0, stream>>>(Wm, Wmt, 1);
        for (int i = 0; i < G.np; i++) {
            const unsigned short* Hpb = Hmpb + (size_t)G.ps[i] * ND;
            wmma_gemm<<<512, 128, 0, stream>>>(Hpb, Wmt, bm, tmp, nullptr, 1, 1.0f, 1);
            mean_kernel<<<DD, 128, 0, stream>>>(tmp, sem + (size_t)i * DD);
        }
        beta_kernel<<<1, 256, 0, stream>>>(sem, qm, G.np, betaW, outF + G.betaOff);
        combine_kernel<<<(ND + 255) / 256, 256, 0, stream>>>(
            Hmp + (size_t)G.ps[0] * ND, Hmp + (size_t)G.ps[1] * ND,
            Hmp + (size_t)G.ps[2] * ND, G.np, betaW, outF + G.outOff);
    }
}
